// GP16_74680891342861
// MI455X (gfx1250) — matrix operand order repaired and checked by reading, not compiled, not run on hardware
//
#include <hip/hip_runtime.h>
#include <hip/hip_bf16.h>

typedef __attribute__((ext_vector_type(16))) __bf16 v16bf;
typedef __attribute__((ext_vector_type(8)))  __bf16 v8bf;
typedef __attribute__((ext_vector_type(8)))  float  v8f;

#define HW   1296      // 36*36
#define CH   512
#define NB   32
#define NKT  81        // 1296 / 16 k-tiles
#define SW   1312      // padded K width = 41*32
#define PI_F 3.14159265358979323846f

// ---------------------------------------------------------------------------
// Kernel 1: L2-normalize x and y over channel dim, emit bf16 in (b, hw, c).
// One wave32 per (which, b, pos) row.
// ---------------------------------------------------------------------------
__global__ void norm_kernel(const float* __restrict__ x,
                            const float* __restrict__ y,
                            __bf16* __restrict__ qn,
                            __bf16* __restrict__ kn) {
    int gw   = (int)((blockIdx.x * blockDim.x + threadIdx.x) >> 5);
    int lane = threadIdx.x & 31;
    const int rows_per_tensor = NB * HW;
    if (gw >= 2 * rows_per_tensor) return;
    int which = (gw >= rows_per_tensor) ? 1 : 0;
    int rowid = which ? (gw - rows_per_tensor) : gw;
    int b   = rowid / HW;
    int pos = rowid % HW;

    const float* src = (which ? y : x) + (size_t)b * CH * HW + pos;
    float vals[16];
    float ss = 0.f;
#pragma unroll
    for (int i = 0; i < 16; ++i) {
        float v = src[(size_t)(i * 32 + lane) * HW];
        vals[i] = v;
        ss += v * v;
    }
#pragma unroll
    for (int off = 16; off; off >>= 1) ss += __shfl_xor(ss, off, 32);
    float inv = 1.0f / fmaxf(sqrtf(ss), 1e-6f);

    __bf16* dst = (which ? kn : qn) + ((size_t)b * HW + pos) * CH;
#pragma unroll
    for (int i = 0; i < 16; ++i) dst[i * 32 + lane] = (__bf16)(vals[i] * inv);
}

// ---------------------------------------------------------------------------
// Kernel 2: fT[d][k] = cos(8*pi*(W[d,0]*gx + W[d,1]*gy + bias[d])), bf16,
// d-major, width padded to SW with zeros.
// ---------------------------------------------------------------------------
__global__ void posenc_kernel(const float* __restrict__ W,
                              const float* __restrict__ bias,
                              __bf16* __restrict__ fT) {
    int d  = blockIdx.x;         // 0..511
    float w0 = W[d * 2 + 0];
    float w1 = W[d * 2 + 1];
    float bb = bias[d];
    for (int k = threadIdx.x; k < SW; k += blockDim.x) {
        float val = 0.f;
        if (k < HW) {
            int iy = k / 36, ix = k % 36;
            float gx = -1.0f + (2.0f * ix + 1.0f) / 36.0f;
            float gy = -1.0f + (2.0f * iy + 1.0f) / 36.0f;
            val = cosf(8.0f * PI_F * (w0 * gx + w1 * gy + bb));
        }
        fT[(size_t)d * SW + k] = (__bf16)val;
    }
}

// ---------------------------------------------------------------------------
// Kernel 3: attention. Block = (q-tile of 16 rows, batch). 256 thr = 8 waves.
// ---------------------------------------------------------------------------
__launch_bounds__(256, 1)
__global__ void attn_kernel(const __bf16* __restrict__ qn,
                            const __bf16* __restrict__ kn,
                            const __bf16* __restrict__ fT,
                            float* __restrict__ out) {
    __shared__ __bf16 Qs[16 * CH];     // 16 KB   q tile (row-major, c contiguous)
    __shared__ float  S[16 * SW];      // 84 KB   raw scores
    __shared__ __bf16 Pbf[16 * SW];    // 42 KB   bf16 softmax probs (WMMA A feed)

    const int qt   = blockIdx.x;       // 0..80
    const int b    = blockIdx.y;       // 0..31
    const int tid  = threadIdx.x;
    const int lane = tid & 31;
    const int wv   = tid >> 5;         // 0..7
    const int row  = lane & 15;        // M row (A) or N col (B/C)
    const int hi   = lane >> 4;        // lane half select

    // ---- async-copy q tile (contiguous 16 KB, 16 B per lane per op) ----
    {
        const char* gsrc = (const char*)(qn + ((size_t)b * HW + qt * 16) * CH);
        unsigned lds_base = (unsigned)(uintptr_t)(&Qs[0]);
#pragma unroll
        for (int i = 0; i < 4; ++i) {
            int idx = tid + i * 256;                       // 16-byte unit
            unsigned    ldsa = lds_base + idx * 16;
            const void* ga   = gsrc + (size_t)idx * 16;
            asm volatile("global_load_async_to_lds_b128 %0, %1, off"
                         :: "v"(ldsa), "v"(ga) : "memory");
        }
        asm volatile("s_wait_asynccnt 0x0" ::: "memory");
    }
    // zero the K-pad columns of the prob buffer
    if (tid < 256) {
        int r = tid >> 4, c = tid & 15;
        Pbf[r * SW + HW + c] = (__bf16)0.0f;
    }
    __syncthreads();

    // ---- phase 1: S[0:16][j*16:+16] = qn_tile . kn_tile^T  (K = 512) ----
    for (int j = wv; j < NKT; j += 8) {
        v8f c = {};
        const __bf16* kp = kn + ((size_t)b * HW + j * 16 + row) * CH + hi * 16;
        const __bf16* qp = Qs + row * CH + hi * 16;
#pragma unroll
        for (int kk = 0; kk < 16; ++kk) {
            union { v16bf v; v8bf h[2]; } a;
            a.h[0] = *(const v8bf*)(qp + kk * 32);        // K = kk*32 + hi*16 + {0..7}
            a.h[1] = *(const v8bf*)(qp + kk * 32 + 8);    // K = kk*32 + hi*16 + {8..15}
            v16bf bf = *(const v16bf*)(kp + kk * 32);     // K = kk*32 + hi*16 + {0..15}
            c = __builtin_amdgcn_wmma_f32_16x16x32_bf16(
                    false, a.v, false, bf, (short)0, c, false, false);
        }
#pragma unroll
        for (int r = 0; r < 8; ++r)
            S[(r + hi * 8) * SW + j * 16 + row] = c[r];
    }
    __syncthreads();

    // ---- phase 2: softmax(S / 0.1) per row -> bf16 probs; 2 rows per wave ----
#pragma unroll
    for (int rr = 0; rr < 2; ++rr) {
        int r = wv * 2 + rr;
        float*  Sr = S + r * SW;
        __bf16* Pr = Pbf + r * SW;
        float m = -3.4e38f;
        for (int k = lane; k < HW; k += 32) m = fmaxf(m, Sr[k]);
#pragma unroll
        for (int off = 16; off; off >>= 1) m = fmaxf(m, __shfl_xor(m, off, 32));
        float sum = 0.f;
        for (int k = lane; k < HW; k += 32) {
            float p = __expf((Sr[k] - m) * 10.0f);   // 1/TEMP = 10
            Sr[k] = p;
            sum += p;
        }
#pragma unroll
        for (int off = 16; off; off >>= 1) sum += __shfl_xor(sum, off, 32);
        float inv = 1.0f / sum;
        for (int k = lane; k < HW; k += 32) Pr[k] = (__bf16)(Sr[k] * inv);
    }
    __syncthreads();

    // ---- phase 3: mu_tile = P . V, each wave owns a 64-wide d slice ----
    v8f acc0 = {}, acc1 = {}, acc2 = {}, acc3 = {};
    const int dbase = wv * 64;
    const __bf16* Pr = Pbf + row * SW + hi * 16;
    const __bf16* vb = fT + (size_t)(dbase + row) * SW + hi * 16;
    for (int kk = 0; kk < 41; ++kk) {
        union { v16bf v; v8bf h[2]; } a;
        a.h[0] = *(const v8bf*)(Pr + kk * 32);        // K+{0..7}
        a.h[1] = *(const v8bf*)(Pr + kk * 32 + 8);    // K+{8..15}
        const __bf16* vp = vb + kk * 32;
        v16bf b0 = *(const v16bf*)(vp);
        v16bf b1 = *(const v16bf*)(vp + 16 * SW);
        v16bf b2 = *(const v16bf*)(vp + 32 * SW);
        v16bf b3 = *(const v16bf*)(vp + 48 * SW);
        acc0 = __builtin_amdgcn_wmma_f32_16x16x32_bf16(false, a.v, false, b0, (short)0, acc0, false, false);
        acc1 = __builtin_amdgcn_wmma_f32_16x16x32_bf16(false, a.v, false, b1, (short)0, acc1, false, false);
        acc2 = __builtin_amdgcn_wmma_f32_16x16x32_bf16(false, a.v, false, b2, (short)0, acc2, false, false);
        acc3 = __builtin_amdgcn_wmma_f32_16x16x32_bf16(false, a.v, false, b3, (short)0, acc3, false, false);
    }

    // ---- epilogue: out[b][d][qt*16 + m]; per-lane 8 consecutive m -> float4 x2 ----
    float* ob = out + (size_t)b * CH * HW + qt * 16 + hi * 8;
    {
        float* o0 = ob + (size_t)(dbase + 0  + row) * HW;
        float* o1 = ob + (size_t)(dbase + 16 + row) * HW;
        float* o2 = ob + (size_t)(dbase + 32 + row) * HW;
        float* o3 = ob + (size_t)(dbase + 48 + row) * HW;
        *(float4*)(o0)     = make_float4(acc0[0], acc0[1], acc0[2], acc0[3]);
        *(float4*)(o0 + 4) = make_float4(acc0[4], acc0[5], acc0[6], acc0[7]);
        *(float4*)(o1)     = make_float4(acc1[0], acc1[1], acc1[2], acc1[3]);
        *(float4*)(o1 + 4) = make_float4(acc1[4], acc1[5], acc1[6], acc1[7]);
        *(float4*)(o2)     = make_float4(acc2[0], acc2[1], acc2[2], acc2[3]);
        *(float4*)(o2 + 4) = make_float4(acc2[4], acc2[5], acc2[6], acc2[7]);
        *(float4*)(o3)     = make_float4(acc3[0], acc3[1], acc3[2], acc3[3]);
        *(float4*)(o3 + 4) = make_float4(acc3[4], acc3[5], acc3[6], acc3[7]);
    }
}

// ---------------------------------------------------------------------------
extern "C" void kernel_launch(void* const* d_in, const int* in_sizes, int n_in,
                              void* d_out, int out_size, void* d_ws, size_t ws_size,
                              hipStream_t stream) {
    const float* x    = (const float*)d_in[0];
    const float* y    = (const float*)d_in[1];
    const float* W    = (const float*)d_in[2];
    const float* bias = (const float*)d_in[3];
    float* out = (float*)d_out;

    // workspace layout: qn | kn | fT  (all bf16)
    __bf16* qn = (__bf16*)d_ws;
    __bf16* kn = qn + (size_t)NB * HW * CH;
    __bf16* fT = kn + (size_t)NB * HW * CH;

    // normalize: one wave per (which, b, pos) row
    {
        int waves   = 2 * NB * HW;               // 82944
        int threads = waves * 32;
        int blocks  = (threads + 255) / 256;     // 10368
        norm_kernel<<<blocks, 256, 0, stream>>>(x, y, qn, kn);
    }
    // positional cosine table (transposed + padded)
    posenc_kernel<<<CH, 256, 0, stream>>>(W, bias, fT);

    // attention: grid = (q-tiles, batch)
    dim3 grid(NKT, NB);
    attn_kernel<<<grid, 256, 0, stream>>>(qn, kn, fT, out);
}
